// MIMICModel_79748952752224
// MI455X (gfx1250) — matrix operand order repaired and checked by reading, not compiled, not run on hardware
//
#include <hip/hip_runtime.h>
#include <hip/hip_bf16.h>
#include <math.h>

// ---------------------------------------------------------------------------
// MI455X (gfx1250, wave32) implementation.
// 16 batch rows per workgroup (16 WGs x 8 wave32), f16 WMMA
// (v_wmma_f32_16x16x32_f16) for every GEMM, fp32 accumulation, weights
// converted+transposed once into f16 workspace (L2-resident, 192MB L2).
// Round-4 change: the opaque (LICM-blocking) weight pointers are returned as
// address-space(1) pointers so weight streaming uses GLOBAL_LOAD_B128
// (LOADcnt only) instead of FLAT_LOAD_B128 (LOADcnt+DScnt) -- decouples the
// L2 weight stream from the LDS A-fragment traffic.
// ---------------------------------------------------------------------------

typedef __attribute__((ext_vector_type(16))) _Float16 v16h;
typedef __attribute__((ext_vector_type(8)))  float    v8f;

// global-address-space f16 (const) for weight streaming
typedef const __attribute__((address_space(1))) _Float16 gf16;

#define DEV __device__ __forceinline__

#if defined(__has_builtin)
#if __has_builtin(__builtin_amdgcn_sched_barrier)
#define SCHED_BARRIER() __builtin_amdgcn_sched_barrier(0)
#endif
#endif
#ifndef SCHED_BARRIER
#define SCHED_BARRIER() asm volatile("" ::: "memory")
#endif

static constexpr int BB = 256;   // batch
static constexpr int TT = 128;   // time steps
static constexpr int DD = 256;   // input dim
static constexpr int HH = 256;   // hidden
static constexpr int NN = 8;     // modules
static constexpr int SS = 32;    // static features

// ---- LDS layout (floats, then halfs) --------------------------------------
static constexpr int RS  = 260;  // fp32 row stride for 256-wide tiles (pad)
static constexpr int KLS = 132;  // fp32 row stride for 128-wide tiles
static constexpr int HSR = 264;  // f16 row stride for 256-wide tiles

static constexpr int F_XT  = 0;
static constexpr int F_VL0 = F_XT  + 16*RS;
static constexpr int F_KL  = F_VL0 + 16*RS;
static constexpr int F_HST = F_KL  + 16*KLS;
static constexpr int F_TA  = F_HST + 16*RS;
static constexpr int F_TB  = F_TA  + 16*RS;
static constexpr int F_SC0 = F_TB  + 16*RS;
static constexpr int F_SC1 = F_SC0 + 128;
static constexpr int F_MP0 = F_SC1 + 128;
static constexpr int F_MP1 = F_MP0 + 128;
static constexpr int F_MSK = F_MP1 + 128;
static constexpr int F_BVF = F_MSK + 128;
static constexpr int F_ATT = F_BVF + 256;
static constexpr int F_TOT = F_ATT + 16*4*8*8;

static constexpr int H_Q   = 0;
static constexpr int H_K   = H_Q  + 16*NN*128;
static constexpr int H_HM  = H_K  + 16*NN*128;
static constexpr int H_VB  = H_HM + NN*16*HSR;
static constexpr int H_TOT = H_VB + NN*16*HSR;

static constexpr size_t SMEM_BYTES = (size_t)F_TOT*4 + (size_t)H_TOT*2; // ~305KB of 320KB

// ---- workspace layout (bytes) ---------------------------------------------
static constexpr size_t WS_WKT   = 0;
static constexpr size_t WS_WVFT  = WS_WKT   + (size_t)128*256*2;
static constexpr size_t WS_WQT   = WS_WVFT  + (size_t)256*256*2;
static constexpr size_t WS_WI2HT = WS_WQT   + (size_t)8*128*256*2;
static constexpr size_t WS_WH2HT = WS_WI2HT + (size_t)8*1024*256*2;
static constexpr size_t WS_WQCT  = WS_WH2HT + (size_t)8*1024*256*2;
static constexpr size_t WS_WKCT  = WS_WQCT  + (size_t)8*128*256*2;
static constexpr size_t WS_WVCT  = WS_WKCT  + (size_t)8*128*256*2;
static constexpr size_t WS_WCOT  = WS_WVCT  + (size_t)8*1024*256*2;
static constexpr size_t WS_HS    = WS_WCOT  + (size_t)8*1024*256*2;
static constexpr size_t WS_CS    = WS_HS    + (size_t)BB*NN*HH*4;

// ---------------------------------------------------------------------------
// Opaque pointer laundering: breaks loop-invariance of weight loads so LICM
// cannot hoist them out of the t-loop, and returns a GLOBAL address-space
// pointer so the loads lower to GLOBAL_LOAD (LOADcnt only), not FLAT.
// ---------------------------------------------------------------------------
DEV gf16* opaque_w(const _Float16* p) {
  unsigned long long v = (unsigned long long)p;
  asm volatile("" : "+s"(v));
  return (gf16*)v;
}

// ---------------------------------------------------------------------------
// WMMA fragment helpers (wave32 layouts per CDNA5 ISA 7.12.2)
// A (16x32 f16): lane L -> row L&15; halves 0..7 -> K = 8*(L>>4)+i,
//                halves 8..15 -> K = 16 + 8*(L>>4)+i   (two contiguous runs)
// B (32x16 f16): lane L -> col L&15; halves 0..7 -> K = 8*(L>>4)+i, 8..15 -> K = 16+8*(L>>4)+i (as A)
// C (16x16 f32): VGPR i -> row 8*(L>>4)+i; lane -> col L&15
// ---------------------------------------------------------------------------
DEV v8f vzero8() { v8f z = {0.f,0.f,0.f,0.f,0.f,0.f,0.f,0.f}; return z; }

DEV v16h fragA_f32(const float* __restrict__ tile, int rs, int kb, int lane) {
  const int r = lane & 15, g = (lane >> 4) * 8;
  const float* p0 = tile + r*rs + kb + g;
  v16h a;
#pragma unroll
  for (int i = 0; i < 8; ++i) { a[i] = (_Float16)p0[i]; a[i+8] = (_Float16)p0[16+i]; }
  return a;
}

DEV v16h fragA_inp(const float* __restrict__ vl0, const float* __restrict__ bvf,
                   int kb, int lane, float s, float s2) {
  const int r = lane & 15, g = (lane >> 4) * 8;
  const float* p0 = vl0 + r*RS + kb + g;
  const float* q0 = bvf + kb + g;
  v16h a;
#pragma unroll
  for (int i = 0; i < 8; ++i) {
    a[i]   = (_Float16)(s*p0[i]    + s2*q0[i]);
    a[i+8] = (_Float16)(s*p0[16+i] + s2*q0[16+i]);
  }
  return a;
}

DEV v16h fragA_h(const _Float16* __restrict__ tile, int rs, int kb, int lane) {
  const int r = lane & 15, g = (lane >> 4) * 8;
  const _Float16* p0 = tile + r*rs + kb + g;
  v16h a;
#pragma unroll
  for (int i = 0; i < 8; ++i) { a[i] = p0[i]; a[i+8] = p0[16+i]; }
  return a;
}

DEV v16h fragB(gf16* wt, int Kdim, int nb, int kb, int lane) {
  gf16* p = wt + (size_t)(nb + (lane & 15)) * (size_t)Kdim
               + (size_t)(kb + ((lane >> 4) << 3));
  v16h b;
#pragma unroll
  for (int i = 0; i < 8; ++i) { b[i] = p[i]; b[i+8] = p[16+i]; }
  return b;
}

// K=256 GEMM chain; sched barriers every 2 K-steps cap in-flight B fragments.
DEV v8f gemm8(const v16h* a, gf16* wt, int Kdim,
              int nb, int kb0, int lane, v8f c) {
#pragma unroll
  for (int kt = 0; kt < 8; ++kt) {
    c = __builtin_amdgcn_wmma_f32_16x16x32_f16(
          false, a[kt], false, fragB(wt, Kdim, nb, kb0 + kt*32, lane),
          (short)0, c, false, false);
    if ((kt & 1) == 1) SCHED_BARRIER();
  }
  return c;
}

DEV v8f gemm_pre(const v16h* ai, const v16h* ah,
                 gf16* Wi, gf16* Wh, int nb, int lane) {
  v8f c = gemm8(ai, Wi, 256, nb, 0, lane, vzero8());
  return gemm8(ah, Wh, 256, nb, 0, lane, c);
}

DEV float sigf(float x) { return 1.f / (1.f + __expf(-x)); }

// ---------------------------------------------------------------------------
// Prologue kernels: weight convert+transpose to f16, Wv fold, state zero
// ---------------------------------------------------------------------------
__global__ void transpose_f16_kernel(const float* __restrict__ src, _Float16* __restrict__ dst,
                                     int Kd, int Od, int cnt) {
  const long long per = (long long)Kd * Od;
  const long long total = per * cnt;
  for (long long idx = (long long)blockIdx.x*blockDim.x + threadIdx.x; idx < total;
       idx += (long long)gridDim.x*blockDim.x) {
    const int m = (int)(idx / per);
    const long long rem = idx - (long long)m * per;
    const int k = (int)(rem / Od);
    const int o = (int)(rem % Od);
    dst[(long long)m*per + (long long)o*Kd + k] = (_Float16)src[idx];
  }
}

__global__ void fold_wv_kernel(const float* __restrict__ Wv, _Float16* __restrict__ dst) {
  const int idx = blockIdx.x*blockDim.x + threadIdx.x;
  if (idx < 256*256) {
    const int o = idx & 255, k = idx >> 8;
    dst[o*256 + k] = (_Float16)(0.5f * (Wv[k*512 + o] + Wv[k*512 + o + 256]));
  }
}

__global__ void zero_f32_kernel(float* __restrict__ p, long long n) {
  for (long long i = (long long)blockIdx.x*blockDim.x + threadIdx.x; i < n;
       i += (long long)gridDim.x*blockDim.x) p[i] = 0.f;
}

// ---------------------------------------------------------------------------
// Main scan kernel: grid = 16 blocks (16 batch rows each), 256 threads (8 waves)
// ---------------------------------------------------------------------------
__global__ __launch_bounds__(256, 2)
void mimic_main(const float* __restrict__ x, const float* __restrict__ bk,
                const float* __restrict__ bv,
                const _Float16* __restrict__ WkTg,  const _Float16* __restrict__ WvFTg,
                const _Float16* __restrict__ WqTg,  const _Float16* __restrict__ Wi2hTg,
                const _Float16* __restrict__ Wh2hTg,const _Float16* __restrict__ WqcTg,
                const _Float16* __restrict__ WkcTg, const _Float16* __restrict__ WvcTg,
                const _Float16* __restrict__ WcoTg,
                float* __restrict__ hsG, float* __restrict__ csG) {
  extern __shared__ char smemRaw[];
  float* fb   = (float*)smemRaw;
  float* xt   = fb + F_XT;
  float* vl0  = fb + F_VL0;
  float* klb  = fb + F_KL;
  float* hstg = fb + F_HST;
  float* tmpA = fb + F_TA;
  float* tmpB = fb + F_TB;
  float* sc0  = fb + F_SC0;
  float* sc1  = fb + F_SC1;
  float* mp0  = fb + F_MP0;
  float* mp1  = fb + F_MP1;
  float* mskv = fb + F_MSK;
  float* bvf  = fb + F_BVF;
  float* attb = fb + F_ATT;
  _Float16* hb   = (_Float16*)(fb + F_TOT);
  _Float16* qsh  = hb + H_Q;
  _Float16* ksh  = hb + H_K;
  _Float16* hmv  = hb + H_HM;
  _Float16* vbuf = hb + H_VB;

  const int tid  = threadIdx.x;
  const int lane = tid & 31;
  const int w    = tid >> 5;            // wave id (wave32)
  const int b0   = blockIdx.x * 16;

  // folded Vl bias (mean over NIH groups), step-invariant
  if (tid < 256) bvf[tid] = 0.5f * (bv[tid] + bv[tid + 256]);
  __syncthreads();

  for (int t = 0; t < TT; ++t) {
    // re-launder weight pointers each step: defeats LICM hoisting of the
    // (t-invariant) weight fragment loads; weights re-stream from L2 via
    // GLOBAL_LOAD (addrspace-1 pointers).
    gf16* WkT   = opaque_w(WkTg);
    gf16* WvFT  = opaque_w(WvFTg);
    gf16* WqT   = opaque_w(WqTg);
    gf16* Wi2hT = opaque_w(Wi2hTg);
    gf16* Wh2hT = opaque_w(Wh2hTg);
    gf16* WqcT  = opaque_w(WqcTg);
    gf16* WkcT  = opaque_w(WkcTg);
    gf16* WvcT  = opaque_w(WvcTg);
    gf16* WcoT  = opaque_w(WcoTg);

    // ---- phase 0: stage x_t tile --------------------------------------
    for (int i = tid; i < 16*256; i += 256) {
      const int r = i >> 8, c2 = i & 255;
      xt[r*RS + c2] = x[((size_t)(b0 + r)*TT + t)*DD + c2];
    }
    if (t + 1 < TT)
      __builtin_prefetch(&x[((size_t)(b0 + (tid & 15))*TT + (t + 1))*DD], 0, 3);
    __syncthreads();

    v16h ax[8];
#pragma unroll
    for (int kt = 0; kt < 8; ++kt) ax[kt] = fragA_f32(xt, RS, kt*32, lane);

    // ---- A1: Kl = x@Wk + bk (16x128, wave w -> col tile w) ------------
    {
      v8f c = gemm8(ax, WkT, 256, w*16, 0, lane, vzero8());
      const int col = w*16 + (lane & 15), mb = (lane >> 4) * 8;
      const float bkv = bk[col];
#pragma unroll
      for (int i = 0; i < 8; ++i) klb[(mb+i)*KLS + col] = c[i] + bkv;
    }
    // ---- A2: Vl0 = x@WvFold + bvf (16x256, 2 tiles per wave) ----------
#pragma unroll
    for (int s = 0; s < 2; ++s) {
      const int nb = (w*2 + s) * 16;
      v8f c = gemm8(ax, WvFT, 256, nb, 0, lane, vzero8());
      const int col = nb + (lane & 15), mb = (lane >> 4) * 8;
      const float bvv = bvf[col];
#pragma unroll
      for (int i = 0; i < 8; ++i) vl0[(mb+i)*RS + col] = c[i] + bvv;
    }
    __syncthreads();

    // ---- A3: per-module Ql and routing scores -------------------------
    for (int n = 0; n < NN; ++n) {
      for (int i = tid; i < 16*256; i += 256) {
        const int r = i >> 8, c2 = i & 255;
        hstg[r*RS + c2] = hsG[(((size_t)(b0 + r))*NN + n)*HH + c2];
      }
      __syncthreads();
      v16h ahh[8];
#pragma unroll
      for (int kt = 0; kt < 8; ++kt) ahh[kt] = fragA_f32(hstg, RS, kt*32, lane);
      {
        v8f c = gemm8(ahh, WqT + (size_t)n*128*256, 256, w*16, 0, lane, vzero8());
        const int col = w*16 + (lane & 15), mb = (lane >> 4) * 8;
#pragma unroll
        for (int i = 0; i < 8; ++i) tmpA[(mb+i)*KLS + col] = c[i];
      }
      __syncthreads();
      if (w == 0 && lane < 16) {   // scores: mean over NIH of dot/sqrt(64) = dot128/16
        float d0 = 0.f, d1 = 0.f;
        for (int j = 0; j < 128; ++j) {
          const float qv = tmpA[lane*KLS + j];
          d0 += qv * klb[lane*KLS + j];
          d1 += qv * bk[j];               // Kl[t=1] row is just bk
        }
        sc0[lane*8 + n] = d0 * (1.f/16.f);
        sc1[lane*8 + n] = d1 * (1.f/16.f);
      }
      __syncthreads();
    }

    // ---- A4: top-4 mask + 2-way softmax over t (wave 0) ---------------
    if (w == 0 && lane < 16) {
      unsigned used = 0;
      for (int k = 0; k < 4; ++k) {
        float best = -3.0e38f; int bi = 0;
        for (int n = 0; n < NN; ++n)
          if (!((used >> n) & 1u)) { const float v = sc0[lane*8 + n]; if (v > best) { best = v; bi = n; } }
        used |= 1u << bi;
      }
      for (int n = 0; n < NN; ++n) {
        const float m  = ((used >> n) & 1u) ? 1.f : 0.f;
        const float p0 = 1.f / (1.f + __expf(sc1[lane*8 + n] - sc0[lane*8 + n]));
        mskv[lane*8 + n] = m;
        mp0[lane*8 + n]  = m * p0;
        mp1[lane*8 + n]  = m * (1.f - p0);
      }
    }
    __syncthreads();

    // ---- B: per-module LSTM (gate chunks: f, i, g, o) -----------------
    for (int n = 0; n < NN; ++n) {
      for (int i = tid; i < 16*256; i += 256) {
        const int r = i >> 8, c2 = i & 255;
        hstg[r*RS + c2] = hsG[(((size_t)(b0 + r))*NN + n)*HH + c2];
      }
      __syncthreads();
      const int rr = lane & 15;
      const float sA = mp0[rr*8 + n], sB = mp1[rr*8 + n];
      v16h ah[8], ai[8];
#pragma unroll
      for (int kt = 0; kt < 8; ++kt) {
        ah[kt] = fragA_f32(hstg, RS, kt*32, lane);
        ai[kt] = fragA_inp(vl0, bvf, kt*32, lane, sA, sB);
      }
      gf16* Wi = Wi2hT + (size_t)n*1024*256;
      gf16* Wh = Wh2hT + (size_t)n*1024*256;
#pragma unroll
      for (int s = 0; s < 2; ++s) {
        const int ct  = w*2 + s;
        const int col = ct*16 + (lane & 15);
        const int mb  = (lane >> 4) * 8;
        v8f c;
        // f gate (cols 256..511): tmpA = f * cs_old
        c = gemm_pre(ai, ah, Wi, Wh, 256 + ct*16, lane);
#pragma unroll
        for (int i = 0; i < 8; ++i) {
          const int r = mb + i;
          tmpA[r*RS + col] = sigf(c[i]) * csG[(((size_t)(b0 + r))*NN + n)*HH + col];
        }
        // i gate (cols 0..255)
        c = gemm_pre(ai, ah, Wi, Wh, ct*16, lane);
#pragma unroll
        for (int i = 0; i < 8; ++i) { const int r = mb + i; tmpB[r*RS + col] = sigf(c[i]); }
        // g (cols 768..1023): c_new = cs*f + i*tanh(g); masked cs update
        c = gemm_pre(ai, ah, Wi, Wh, 768 + ct*16, lane);
#pragma unroll
        for (int i = 0; i < 8; ++i) {
          const int r = mb + i;
          const float cn = tmpA[r*RS + col] + tmpB[r*RS + col] * tanhf(c[i]);
          tmpA[r*RS + col] = cn;
          if (mskv[r*8 + n] != 0.f) csG[(((size_t)(b0 + r))*NN + n)*HH + col] = cn;
        }
        // o gate (cols 512..767): hm = mask * o * tanh(c_new)
        c = gemm_pre(ai, ah, Wi, Wh, 512 + ct*16, lane);
#pragma unroll
        for (int i = 0; i < 8; ++i) {
          const int r = mb + i;
          const float hv = sigf(c[i]) * tanhf(tmpA[r*RS + col]);
          hmv[(n*16 + r)*HSR + col] = (_Float16)(mskv[r*8 + n] * hv);
        }
      }
      __syncthreads();
    }

    // ---- C1: q,k projections (wave w handles module n=w) --------------
    {
      const int n = w;
      const _Float16* hmT = hmv + n*16*HSR;
      v16h am[8];
#pragma unroll
      for (int kt = 0; kt < 8; ++kt) am[kt] = fragA_h(hmT, HSR, kt*32, lane);
      gf16* Wq_ = WqcT + (size_t)n*128*256;
      gf16* Wk_ = WkcT + (size_t)n*128*256;
#pragma unroll
      for (int ntb = 0; ntb < 8; ++ntb) {
        const int col = ntb*16 + (lane & 15), mb = (lane >> 4) * 8;
        v8f c = gemm8(am, Wq_, 256, ntb*16, 0, lane, vzero8());
#pragma unroll
        for (int i = 0; i < 8; ++i) qsh[((mb+i)*NN + n)*128 + col] = (_Float16)c[i];
        c = gemm8(am, Wk_, 256, ntb*16, 0, lane, vzero8());
#pragma unroll
        for (int i = 0; i < 8; ++i) ksh[((mb+i)*NN + n)*128 + col] = (_Float16)c[i];
      }
    }
    __syncthreads();

    // ---- C2: attention logits + softmax over modules ------------------
    {
      const int n = w;
      for (int p = lane; p < 64; p += 32) {
        const int r = p >> 2, h = p & 3;
        float lg[8]; float mx = -3.0e38f;
        for (int m = 0; m < NN; ++m) {
          float d = 0.f;
#pragma unroll
          for (int j = 0; j < 32; ++j)
            d += (float)qsh[(r*NN + n)*128 + h*32 + j] * (float)ksh[(r*NN + m)*128 + h*32 + j];
          d *= 0.1767766952966369f;     // 1/sqrt(32)
          lg[m] = d; mx = fmaxf(mx, d);
        }
        float ssum = 0.f;
        for (int m = 0; m < NN; ++m) { lg[m] = __expf(lg[m] - mx); ssum += lg[m]; }
        const float sc = mskv[r*8 + n] / ssum;
        for (int m = 0; m < NN; ++m) attb[((r*4 + h)*8 + n)*8 + m] = lg[m] * sc;
      }
    }
    __syncthreads();

    // ---- C3: v projections, att-weighted ctx, ctx@Wco + hm ------------
    // acc[16] (128 VGPRs) stays register-resident across the head loop;
    // kt-outer ordering keeps only ONE ctx A-fragment live at a time.
    {
      const int n = w;
      const int r = lane & 15, g = (lane >> 4) * 8;
      v8f acc[16];
#pragma unroll
      for (int q2 = 0; q2 < 16; ++q2) acc[q2] = vzero8();

      for (int h = 0; h < 4; ++h) {
        { // v_h for m = w (cooperative across waves)
          const int m = w;
          const _Float16* hmT = hmv + m*16*HSR;
          v16h am[8];
#pragma unroll
          for (int kt = 0; kt < 8; ++kt) am[kt] = fragA_h(hmT, HSR, kt*32, lane);
          gf16* Wv_ = WvcT + (size_t)m*1024*256;
#pragma unroll
          for (int ntb = 0; ntb < 16; ++ntb) {
            v8f c = gemm8(am, Wv_, 256, h*256 + ntb*16, 0, lane, vzero8());
            const int col = ntb*16 + (lane & 15), mb = (lane >> 4) * 8;
#pragma unroll
            for (int i = 0; i < 8; ++i) vbuf[(m*16 + mb + i)*HSR + col] = (_Float16)c[i];
          }
        }
        __syncthreads();

        float av[8];
#pragma unroll
        for (int m = 0; m < NN; ++m) av[m] = attb[((r*4 + h)*8 + n)*8 + m];
        gf16* Wc_ = WcoT + (size_t)n*256*1024;
#pragma unroll
        for (int kt = 0; kt < 8; ++kt) {
          v16h a;   // single live ctx fragment (att-weighted sum over modules)
#pragma unroll
          for (int i = 0; i < 8; ++i) {
            const int k0 = kt*32 + g + i, k1 = kt*32 + 16 + g + i;
            float u0 = 0.f, u1 = 0.f;
#pragma unroll
            for (int m = 0; m < NN; ++m) {
              u0 += av[m] * (float)vbuf[(m*16 + r)*HSR + k0];
              u1 += av[m] * (float)vbuf[(m*16 + r)*HSR + k1];
            }
            a[i] = (_Float16)u0; a[i+8] = (_Float16)u1;
          }
          SCHED_BARRIER();
#pragma unroll
          for (int ntb = 0; ntb < 16; ++ntb) {
            acc[ntb] = __builtin_amdgcn_wmma_f32_16x16x32_f16(
                false, a, false, fragB(Wc_, 1024, ntb*16, h*256 + kt*32, lane),
                (short)0, acc[ntb], false, false);
            if ((ntb & 3) == 3) SCHED_BARRIER();
          }
        }
        __syncthreads();   // vbuf fully consumed; next h may overwrite it
      }

      // epilogue: ctx += hm; hs2 = mask ? ctx : hs (masked write only)
#pragma unroll
      for (int ntb = 0; ntb < 16; ++ntb) {
        const int col = ntb*16 + (lane & 15), mb = (lane >> 4) * 8;
#pragma unroll
        for (int i = 0; i < 8; ++i) {
          const int rr2 = mb + i;
          if (mskv[rr2*8 + n] != 0.f)
            hsG[(((size_t)(b0 + rr2))*NN + n)*HH + col] =
                acc[ntb][i] + (float)hmv[(n*16 + rr2)*HSR + col];
        }
      }
    }
    __threadfence();
    __syncthreads();
  } // t
}

// ---------------------------------------------------------------------------
// Head: out[b] = ((concat(hs[b], static[b]) @ W1 + b1) @ W2 + b2)
// ---------------------------------------------------------------------------
__global__ void head_kernel(const float* __restrict__ hsG, const float* __restrict__ statp,
                            const float* __restrict__ W1, const float* __restrict__ b1,
                            const float* __restrict__ W2, const float* __restrict__ b2,
                            float* __restrict__ out) {
  const int b = threadIdx.x;
  if (b >= BB) return;
  float acc[10];
#pragma unroll
  for (int j = 0; j < 10; ++j) acc[j] = b1[j];
  for (int k = 0; k < NN*HH; ++k) {
    const float v = hsG[(size_t)b*NN*HH + k];
#pragma unroll
    for (int j = 0; j < 10; ++j) acc[j] += v * W1[k*10 + j];
  }
  for (int k = 0; k < SS; ++k) {
    const float v = statp[b*SS + k];
#pragma unroll
    for (int j = 0; j < 10; ++j) acc[j] += v * W1[(NN*HH + k)*10 + j];
  }
  float o = b2[0];
#pragma unroll
  for (int j = 0; j < 10; ++j) o += acc[j] * W2[j];
  out[b] = o;
}

// ---------------------------------------------------------------------------
extern "C" void kernel_launch(void* const* d_in, const int* in_sizes, int n_in,
                              void* d_out, int out_size, void* d_ws, size_t ws_size,
                              hipStream_t stream) {
  (void)in_sizes; (void)n_in; (void)out_size; (void)ws_size;
  const float* x    = (const float*)d_in[0];
  const float* stat = (const float*)d_in[1];
  const float* Wk   = (const float*)d_in[2];
  const float* bk   = (const float*)d_in[3];
  const float* Wv   = (const float*)d_in[4];
  const float* bv   = (const float*)d_in[5];
  const float* Wq   = (const float*)d_in[6];
  const float* Wi2h = (const float*)d_in[7];
  const float* Wh2h = (const float*)d_in[8];
  const float* Wqc  = (const float*)d_in[9];
  const float* Wkc  = (const float*)d_in[10];
  const float* Wvc  = (const float*)d_in[11];
  const float* Wco  = (const float*)d_in[12];
  const float* W1   = (const float*)d_in[13];
  const float* b1   = (const float*)d_in[14];
  const float* W2   = (const float*)d_in[15];
  const float* b2   = (const float*)d_in[16];
  float* out = (float*)d_out;

  char* ws = (char*)d_ws;
  _Float16* WkT   = (_Float16*)(ws + WS_WKT);
  _Float16* WvFT  = (_Float16*)(ws + WS_WVFT);
  _Float16* WqT   = (_Float16*)(ws + WS_WQT);
  _Float16* Wi2hT = (_Float16*)(ws + WS_WI2HT);
  _Float16* Wh2hT = (_Float16*)(ws + WS_WH2HT);
  _Float16* WqcT  = (_Float16*)(ws + WS_WQCT);
  _Float16* WkcT  = (_Float16*)(ws + WS_WKCT);
  _Float16* WvcT  = (_Float16*)(ws + WS_WVCT);
  _Float16* WcoT  = (_Float16*)(ws + WS_WCOT);
  float* hsG = (float*)(ws + WS_HS);
  float* csG = (float*)(ws + WS_CS);

  auto blocksFor = [](long long total) {
    long long g = (total + 255) / 256;
    if (g > 4096) g = 4096;
    return dim3((unsigned)g);
  };

  // prologue: convert + transpose weights into f16 workspace (L2-resident)
  transpose_f16_kernel<<<blocksFor(256LL*128),      256, 0, stream>>>(Wk,   WkT,   256, 128,  1);
  fold_wv_kernel      <<<blocksFor(256LL*256),      256, 0, stream>>>(Wv,   WvFT);
  transpose_f16_kernel<<<blocksFor(8LL*256*128),    256, 0, stream>>>(Wq,   WqT,   256, 128,  8);
  transpose_f16_kernel<<<blocksFor(8LL*256*1024),   256, 0, stream>>>(Wi2h, Wi2hT, 256, 1024, 8);
  transpose_f16_kernel<<<blocksFor(8LL*256*1024),   256, 0, stream>>>(Wh2h, Wh2hT, 256, 1024, 8);
  transpose_f16_kernel<<<blocksFor(8LL*256*128),    256, 0, stream>>>(Wqc,  WqcT,  256, 128,  8);
  transpose_f16_kernel<<<blocksFor(8LL*256*128),    256, 0, stream>>>(Wkc,  WkcT,  256, 128,  8);
  transpose_f16_kernel<<<blocksFor(8LL*256*1024),   256, 0, stream>>>(Wvc,  WvcT,  256, 1024, 8);
  transpose_f16_kernel<<<blocksFor(8LL*1024*256),   256, 0, stream>>>(Wco,  WcoT,  1024, 256, 8);
  zero_f32_kernel     <<<blocksFor((long long)BB*NN*HH*2), 256, 0, stream>>>(hsG, (long long)BB*NN*HH*2);

  (void)hipFuncSetAttribute((const void*)mimic_main,
                            hipFuncAttributeMaxDynamicSharedMemorySize, (int)SMEM_BYTES);

  mimic_main<<<dim3(BB/16), dim3(256), SMEM_BYTES, stream>>>(
      x, bk, bv, WkT, WvFT, WqT, Wi2hT, Wh2hT, WqcT, WkcT, WvcT, WcoT, hsG, csG);

  head_kernel<<<dim3(1), dim3(256), 0, stream>>>(hsG, stat, W1, b1, W2, b2, out);
}
